// EdgeClassifier_27917287424275
// MI455X (gfx1250) — matrix operand order repaired and checked by reading, not compiled, not run on hardware
//
#include <hip/hip_runtime.h>
#include <hip/hip_bf16.h>

typedef __attribute__((ext_vector_type(16))) _Float16 v16h;
typedef __attribute__((ext_vector_type(8)))  float    v8f;

union Frag { v16h h; unsigned u32[8]; };

static __device__ __forceinline__ int kpat(int j, int half) {
    // 16-bit A/B layout: pair j of lane-half -> K offset within a 32-wide chunk
    return ((j >> 2) << 4) | (half << 3) | ((j & 3) << 1);
}

static __device__ __forceinline__ unsigned pack2(float a, float b) {
    union { _Float16 h[2]; unsigned u; } t;
    t.h[0] = (_Float16)a; t.h[1] = (_Float16)b;
    return t.u;
}

// ---------------------------------------------------------------------------
// Weight prep: build frag-packed f16 [128x64] matrices:
//   mat 0: [W_rev1 ; root1]   (user,  layer1)   coeff1[1]
//   mat 1: [W_pays1; root1]   (merch, layer1)   coeff1[0]
//   mat 2: [W_rev2 ; root2]   (user,  layer2)   coeff2[1]
//   mat 3: [W_pays2; root2]   (merch, layer2)   coeff2[0]
//   mat 4: w_mlp1
// Packed order: [kc(4)][nt(4)][lane(32)][j(8)] as u32 (two f16: K, K+1)
// ---------------------------------------------------------------------------
__global__ void prep_weights(const float* __restrict__ bases1, const float* __restrict__ coeff1,
                             const float* __restrict__ root1,
                             const float* __restrict__ bases2, const float* __restrict__ coeff2,
                             const float* __restrict__ root2,
                             const float* __restrict__ w_mlp1,
                             unsigned* __restrict__ wfrag) {
    int t = blockIdx.x * 256 + threadIdx.x;
    if (t >= 5 * 4096) return;
    int mat = t >> 12, idx = t & 4095;
    int j = idx & 7, lane = (idx >> 3) & 31, nt = (idx >> 8) & 3, kc = idx >> 10;
    int n = lane & 15, half = lane >> 4;
    int K = kc * 32 + kpat(j, half);
    int col = nt * 16 + n;

    auto getw = [&](int k) -> float {
        if (mat == 4) return w_mlp1[k * 64 + col];
        const float* bases = (mat < 2) ? bases1 : bases2;
        const float* coeff = (mat < 2) ? coeff1 : coeff2;
        const float* root  = (mat < 2) ? root1  : root2;
        if (k < 64) {
            int type = (mat & 1) ? 0 : 1;   // even mats = user(rev)=coeff[1], odd = merch(pays)=coeff[0]
            float s = 0.f;
            #pragma unroll
            for (int b = 0; b < 3; b++)
                s += coeff[type * 3 + b] * bases[(size_t)b * 4096 + k * 64 + col];
            return s;
        }
        return root[(k - 64) * 64 + col];
    };
    wfrag[(size_t)mat * 4096 + idx] = pack2(getw(K), getw(K + 1));
}

// ---------------------------------------------------------------------------
__global__ void zero_kernel(float* __restrict__ p, int n) {
    int i = blockIdx.x * 256 + threadIdx.x;
    if (i < n) p[i] = 0.f;
}

__global__ void degree_kernel(const int* __restrict__ u_idx, const int* __restrict__ m_idx,
                              float* __restrict__ deg_u, float* __restrict__ deg_m, int nedges) {
    int e = blockIdx.x * 256 + threadIdx.x;
    if (e >= nedges) return;
    atomicAdd(&deg_u[u_idx[e]], 1.0f);
    atomicAdd(&deg_m[m_idx[e]], 1.0f);
}

// scatter: agg_m += x_u[u], agg_u += x_m[m]  (16 threads per edge, 4 floats each)
template <typename XT>
__global__ void scatter_kernel(const XT* __restrict__ xu, const XT* __restrict__ xm,
                               const int* __restrict__ u_idx, const int* __restrict__ m_idx,
                               float* __restrict__ agg_u, float* __restrict__ agg_m, int nedges) {
    long long tid = (long long)blockIdx.x * 256 + threadIdx.x;
    int e = (int)(tid >> 4);
    if (e >= nedges) return;
    int g = ((int)tid & 15) * 4;
    int u = u_idx[e], mm = m_idx[e];
    const XT* ru = xu + (size_t)u  * 64 + g;
    const XT* rm = xm + (size_t)mm * 64 + g;
    float* au = agg_u + (size_t)u  * 64 + g;
    float* am = agg_m + (size_t)mm * 64 + g;
    #pragma unroll
    for (int i = 0; i < 4; i++) {
        atomicAdd(am + i, (float)ru[i]);
        atomicAdd(au + i, (float)rm[i]);
    }
}

// ---------------------------------------------------------------------------
// h = act([agg/deg | x] @ W'(128x64) + bias), out in f16.
// One wave per 16-node tile; K=128 as 4 chunks; 4 N-tiles -> 16 WMMAs/wave.
// ---------------------------------------------------------------------------
template <typename XT>
__global__ void transform_kernel(const float* __restrict__ agg, const float* __restrict__ deg,
                                 const XT* __restrict__ x, const unsigned* __restrict__ wfrag,
                                 const float* __restrict__ bias, _Float16* __restrict__ out,
                                 int nnodes, int dorelu) {
    __shared__ unsigned wlds[4096];
    for (int i = threadIdx.x; i < 4096; i += 256) wlds[i] = wfrag[i];
    __syncthreads();

    int wave = threadIdx.x >> 5, lane = threadIdx.x & 31;
    int n = lane & 15, half = lane >> 4;
    int tile0 = (blockIdx.x * 8 + wave) * 16;
    int nrow = tile0 + n;
    int srow = (nrow < nnodes) ? nrow : 0;

    float invd = 1.0f / fmaxf(deg[srow], 1.0f);
    const float* arow = agg + (size_t)srow * 64;
    const XT*    xrow = x   + (size_t)srow * 64;

    Frag a[4];
    #pragma unroll
    for (int kc = 0; kc < 2; kc++)
        #pragma unroll
        for (int j = 0; j < 8; j++) {
            int k = kc * 32 + kpat(j, half);
            a[kc].u32[j] = pack2(arow[k] * invd, arow[k + 1] * invd);
        }
    #pragma unroll
    for (int kc = 2; kc < 4; kc++)
        #pragma unroll
        for (int j = 0; j < 8; j++) {
            int k = (kc - 2) * 32 + kpat(j, half);
            if constexpr (sizeof(XT) == 2)
                a[kc].u32[j] = ((const unsigned*)xrow)[k >> 1];
            else
                a[kc].u32[j] = pack2((float)xrow[k], (float)xrow[k + 1]);
        }

    #pragma unroll
    for (int nt = 0; nt < 4; nt++) {
        v8f acc = {0.f, 0.f, 0.f, 0.f, 0.f, 0.f, 0.f, 0.f};
        #pragma unroll
        for (int kc = 0; kc < 4; kc++) {
            Frag b;
            const unsigned* src = &wlds[((kc * 4 + nt) * 32 + lane) * 8];
            #pragma unroll
            for (int j = 0; j < 8; j++) b.u32[j] = src[j];
            acc = __builtin_amdgcn_wmma_f32_16x16x32_f16(false, a[kc].h, false, b.h,
                                                         (short)0, acc, false, false);
        }
        float bb = bias[nt * 16 + n];
        #pragma unroll
        for (int r = 0; r < 8; r++) {
            int node = tile0 + r + 8 * half;
            float v = acc[r] + bb;
            if (dorelu) v = fmaxf(v, 0.f);
            if (node < nnodes) out[(size_t)node * 64 + nt * 16 + n] = (_Float16)v;
        }
    }
}

// ---------------------------------------------------------------------------
// Edge classifier: z = [h_u[u] | h_m[m]] (f16, K=128); t = relu(z@W1+b1) via
// WMMA; stage 16x64 tile in padded LDS; lane (row,c) does the 64->2 dot.
// ---------------------------------------------------------------------------
__global__ void edge_mlp_kernel(const _Float16* __restrict__ h2u, const _Float16* __restrict__ h2m,
                                const int* __restrict__ u_idx, const int* __restrict__ m_idx,
                                const unsigned* __restrict__ wfrag,
                                const float* __restrict__ b_mlp1,
                                const float* __restrict__ w_mlp2, const float* __restrict__ b_mlp2,
                                float* __restrict__ out, int nedges) {
    __shared__ unsigned wlds[4096];
    __shared__ float tile[8][16][68];   // padded: stride 68 avoids bank conflicts
    for (int i = threadIdx.x; i < 4096; i += 256) wlds[i] = wfrag[i];
    __syncthreads();

    int wave = threadIdx.x >> 5, lane = threadIdx.x & 31;
    int n = lane & 15, half = lane >> 4;
    int e0 = (blockIdx.x * 8 + wave) * 16;
    int e = e0 + n;
    if (e >= nedges) e = nedges - 1;   // clamp (2M % 128 == 0 so never taken)
    int u = u_idx[e], mm = m_idx[e];
    const unsigned* rowu = (const unsigned*)(h2u + (size_t)u  * 64);
    const unsigned* rowm = (const unsigned*)(h2m + (size_t)mm * 64);

    Frag a[4];
    #pragma unroll
    for (int kc = 0; kc < 4; kc++) {
        const unsigned* row = (kc < 2) ? rowu : rowm;
        int kb = (kc & 1) * 32;
        #pragma unroll
        for (int j = 0; j < 8; j++) a[kc].u32[j] = row[(kb + kpat(j, half)) >> 1];
    }

    #pragma unroll
    for (int nt = 0; nt < 4; nt++) {
        v8f acc = {0.f, 0.f, 0.f, 0.f, 0.f, 0.f, 0.f, 0.f};
        #pragma unroll
        for (int kc = 0; kc < 4; kc++) {
            Frag b;
            const unsigned* src = &wlds[((kc * 4 + nt) * 32 + lane) * 8];
            #pragma unroll
            for (int j = 0; j < 8; j++) b.u32[j] = src[j];
            acc = __builtin_amdgcn_wmma_f32_16x16x32_f16(false, a[kc].h, false, b.h,
                                                         (short)0, acc, false, false);
        }
        float bb = b_mlp1[nt * 16 + n];
        #pragma unroll
        for (int r = 0; r < 8; r++)
            tile[wave][r + 8 * half][nt * 16 + n] = fmaxf(acc[r] + bb, 0.f);
    }
    __syncthreads();

    // 64 -> 2: lane handles (row = lane&15, class c = lane>>4)
    int row = lane & 15, c = half;
    float accv = b_mlp2[c];
    #pragma unroll 8
    for (int h = 0; h < 64; h++) accv += tile[wave][row][h] * w_mlp2[h * 2 + c];
    int eo = e0 + row;
    if (eo < nedges) out[(size_t)eo * 2 + c] = accv;
}

// ---------------------------------------------------------------------------
extern "C" void kernel_launch(void* const* d_in, const int* in_sizes, int n_in,
                              void* d_out, int out_size, void* d_ws, size_t ws_size,
                              hipStream_t stream) {
    const float* x_user  = (const float*)d_in[0];
    const float* x_merch = (const float*)d_in[1];
    const float* bases1  = (const float*)d_in[2];
    const float* coeff1  = (const float*)d_in[3];
    const float* root1   = (const float*)d_in[4];
    const float* b1      = (const float*)d_in[5];
    const float* bases2  = (const float*)d_in[6];
    const float* coeff2  = (const float*)d_in[7];
    const float* root2   = (const float*)d_in[8];
    const float* b2      = (const float*)d_in[9];
    const float* w_mlp1  = (const float*)d_in[10];
    const float* b_mlp1  = (const float*)d_in[11];
    const float* w_mlp2  = (const float*)d_in[12];
    const float* b_mlp2  = (const float*)d_in[13];
    const int*   eidx    = (const int*)d_in[14];

    const int NU = in_sizes[0] / 64;       // 200000
    const int NM = in_sizes[1] / 64;       // 50000
    const int NE = in_sizes[14] / 2;       // 2000000
    const int* u_idx = eidx;
    const int* m_idx = eidx + NE;

    auto align256 = [](size_t v) { return (v + 255) & ~(size_t)255; };
    char* w = (char*)d_ws;
    size_t off = 0;
    unsigned* wfrag = (unsigned*)(w + off); off += align256(5 * 4096 * sizeof(unsigned));
    float* deg_u = (float*)(w + off); off += align256((size_t)NU * 4);
    float* deg_m = (float*)(w + off); off += align256((size_t)NM * 4);
    float* agg_u = (float*)(w + off); off += align256((size_t)NU * 64 * 4);
    float* agg_m = (float*)(w + off); off += align256((size_t)NM * 64 * 4);
    _Float16* h1u = (_Float16*)(w + off); off += align256((size_t)NU * 64 * 2);
    _Float16* h1m = (_Float16*)(w + off); off += align256((size_t)NM * 64 * 2);
    _Float16* h2u = (_Float16*)(w + off); off += align256((size_t)NU * 64 * 2);
    _Float16* h2m = (_Float16*)(w + off); off += align256((size_t)NM * 64 * 2);

    const int B = 256;
    auto cdiv = [](long long a, long long b) { return (int)((a + b - 1) / b); };

    // 0) weights (frag-packed f16)
    prep_weights<<<cdiv(5 * 4096, B), B, 0, stream>>>(bases1, coeff1, root1,
                                                      bases2, coeff2, root2, w_mlp1, wfrag);
    // 1) degrees (shared by both layers)
    zero_kernel<<<cdiv(NU, B), B, 0, stream>>>(deg_u, NU);
    zero_kernel<<<cdiv(NM, B), B, 0, stream>>>(deg_m, NM);
    degree_kernel<<<cdiv(NE, B), B, 0, stream>>>(u_idx, m_idx, deg_u, deg_m, NE);

    // 2) layer 1 aggregate + transform (relu)
    zero_kernel<<<cdiv((long long)NU * 64, B), B, 0, stream>>>(agg_u, NU * 64);
    zero_kernel<<<cdiv((long long)NM * 64, B), B, 0, stream>>>(agg_m, NM * 64);
    scatter_kernel<float><<<cdiv((long long)NE * 16, B), B, 0, stream>>>(
        x_user, x_merch, u_idx, m_idx, agg_u, agg_m, NE);
    transform_kernel<float><<<cdiv(cdiv(NU, 16), 8), B, 0, stream>>>(
        agg_u, deg_u, x_user, wfrag + 0 * 4096, b1, h1u, NU, 1);
    transform_kernel<float><<<cdiv(cdiv(NM, 16), 8), B, 0, stream>>>(
        agg_m, deg_m, x_merch, wfrag + 1 * 4096, b1, h1m, NM, 1);

    // 3) layer 2 aggregate + transform (no relu)
    zero_kernel<<<cdiv((long long)NU * 64, B), B, 0, stream>>>(agg_u, NU * 64);
    zero_kernel<<<cdiv((long long)NM * 64, B), B, 0, stream>>>(agg_m, NM * 64);
    scatter_kernel<_Float16><<<cdiv((long long)NE * 16, B), B, 0, stream>>>(
        h1u, h1m, u_idx, m_idx, agg_u, agg_m, NE);
    transform_kernel<_Float16><<<cdiv(cdiv(NU, 16), 8), B, 0, stream>>>(
        agg_u, deg_u, h1u, wfrag + 2 * 4096, b2, h2u, NU, 0);
    transform_kernel<_Float16><<<cdiv(cdiv(NM, 16), 8), B, 0, stream>>>(
        agg_m, deg_m, h1m, wfrag + 3 * 4096, b2, h2m, NM, 0);

    // 4) edge classifier
    edge_mlp_kernel<<<cdiv(cdiv(NE, 16), 8), B, 0, stream>>>(
        h2u, h2m, u_idx, m_idx, wfrag + 4 * 4096, b_mlp1, w_mlp2, b_mlp2,
        (float*)d_out, NE);
}
